// LSTMCell_9972914061861
// MI455X (gfx1250) — compile-verified, matrix operand order checked by reading, not run on hardware
//
#include <hip/hip_runtime.h>
#include <hip/hip_bf16.h>

typedef __bf16 bf16;
typedef __attribute__((ext_vector_type(16))) __bf16 v16bf;
typedef __attribute__((ext_vector_type(8)))  __bf16 v8bf;
typedef __attribute__((ext_vector_type(8)))  float  v8f;

#define T_DIM 512
#define B_DIM 64
#define I_DIM 1024
#define H_DIM 1024
#define G_DIM 4096   // 4*H

// ---------------------------------------------------------------------------
// WMMA bf16 fragment loader (16x32 tile, A-layout; B fed as W rows == B^T).
// ISA 16-bit A layout: lanes 0-15 -> row r, K {0..7, 16..23}; lanes 16-31 ->
// row r, K {8..15, 24..31}.  Two 16B loads per lane.
// ---------------------------------------------------------------------------
__device__ __forceinline__ v16bf load_frag16x32(const bf16* __restrict__ base,
                                                int row0, int ld, int k0, int lane) {
  const int half = lane >> 4;
  const int r    = lane & 15;
  const bf16* p  = base + (size_t)(row0 + r) * ld + k0 + half * 8;
  v8bf lo = *(const v8bf*)(p);        // K = k0 + 8h .. +7
  v8bf hi = *(const v8bf*)(p + 16);   // K = k0 + 16 + 8h .. +7
  return __builtin_shufflevector(lo, hi, 0,1,2,3,4,5,6,7,8,9,10,11,12,13,14,15);
}

// ---------------------------------------------------------------------------
// Prep kernels: fp32 -> bf16 conversion, bias fold, state init
// ---------------------------------------------------------------------------
__global__ void cvt_f32_to_bf16(const float* __restrict__ s, bf16* __restrict__ d, int n) {
  int i = blockIdx.x * blockDim.x + threadIdx.x;
  int stride = gridDim.x * blockDim.x;
  for (; i < n; i += stride) d[i] = (bf16)s[i];
}

__global__ void prep_bias_state(const float* __restrict__ b_ih, const float* __restrict__ b_hh,
                                float* __restrict__ bias,
                                const float* __restrict__ h0, const float* __restrict__ c0,
                                bf16* __restrict__ hbuf0, float* __restrict__ cstate) {
  int i = blockIdx.x * blockDim.x + threadIdx.x;
  int stride = gridDim.x * blockDim.x;
  for (int j = i; j < G_DIM; j += stride) bias[j] = b_ih[j] + b_hh[j];
  for (int j = i; j < B_DIM * H_DIM; j += stride) {
    cstate[j] = c0[j];
    hbuf0[j]  = (bf16)h0[j];
  }
}

// ---------------------------------------------------------------------------
// Phase 1: x_pre[T*B, 4H] = X_bf16 [T*B, I] * W_ih_bf16^T [I, 4H] + bias
// Block = 64x64 tile, 8 waves, each wave: 1 M-tile x 2 N-tiles.
// ---------------------------------------------------------------------------
__global__ void __launch_bounds__(256) xpre_gemm(const bf16* __restrict__ X,
                                                 const bf16* __restrict__ W,
                                                 const float* __restrict__ bias,
                                                 float* __restrict__ xpre) {
  const int tid  = threadIdx.x;
  const int lane = tid & 31;
  const int wid  = tid >> 5;
  const int m0 = blockIdx.y * 64 + (wid & 3) * 16;
  const int n0 = blockIdx.x * 64 + (wid >> 2) * 32;
  const int n1 = n0 + 16;

  v8f acc0 = {}; v8f acc1 = {};
#pragma unroll 2
  for (int k = 0; k < I_DIM; k += 32) {
    v16bf a  = load_frag16x32(X, m0, I_DIM, k, lane);
    v16bf b0 = load_frag16x32(W, n0, I_DIM, k, lane);
    v16bf b1 = load_frag16x32(W, n1, I_DIM, k, lane);
    acc0 = __builtin_amdgcn_wmma_f32_16x16x32_bf16(false, a, false, b0, (short)0, acc0, false, false);
    acc1 = __builtin_amdgcn_wmma_f32_16x16x32_bf16(false, a, false, b1, (short)0, acc1, false, false);
  }

  const int half = lane >> 4;
  const int col  = lane & 15;
  const float bv0 = bias[n0 + col];
  const float bv1 = bias[n1 + col];
#pragma unroll
  for (int v = 0; v < 8; ++v) {
    const int row = m0 + v + 8 * half;
    xpre[(size_t)row * G_DIM + n0 + col] = acc0[v] + bv0;
    xpre[(size_t)row * G_DIM + n1 + col] = acc1[v] + bv1;
  }
}

// ---------------------------------------------------------------------------
// Phase 2: one recurrent step.
// Grid (H/16, 2), block = 128 threads (4 waves).  Block (bx,my) owns hidden
// cols [bx*16,+16) for ALL 4 gates and batch rows [my*32,+32).
// Wave w == gate w; each wave computes BOTH 16-row M-tiles of its gate slab,
// reusing one W_hh B-fragment per k-chunk:
//   per k-iter: 2 A-loads + 1 B-load -> 2 independent WMMAs  (2x ILP).
// GEMM results + x_pre land in LDS, then fused gate math updates c/h.
// ---------------------------------------------------------------------------
__global__ void __launch_bounds__(128) lstm_step(const bf16* __restrict__ Whh,
                                                 const bf16* __restrict__ hprev,
                                                 bf16* __restrict__ hnext,
                                                 float* __restrict__ cstate,
                                                 const float* __restrict__ xpre_t,
                                                 float* __restrict__ out_t,
                                                 float* __restrict__ hf,
                                                 float* __restrict__ cf,
                                                 int is_last) {
  __shared__ float pre[4][32][16];

  const int tid  = threadIdx.x;
  const int lane = tid & 31;
  const int gw   = tid >> 5;                       // wave id == gate 0..3 (i,f,o,g)
  const int m0a  = blockIdx.y * 32;                // batch rows 0..15 of slab
  const int m0b  = m0a + 16;                       // batch rows 16..31 of slab
  const int n0   = gw * H_DIM + blockIdx.x * 16;   // W_hh row base (gate space)

  v8f acc0 = {}; v8f acc1 = {};
#pragma unroll 2
  for (int k = 0; k < H_DIM; k += 32) {
    v16bf a0 = load_frag16x32(hprev, m0a, H_DIM, k, lane);
    v16bf a1 = load_frag16x32(hprev, m0b, H_DIM, k, lane);
    v16bf b  = load_frag16x32(Whh,   n0,  H_DIM, k, lane);
    acc0 = __builtin_amdgcn_wmma_f32_16x16x32_bf16(false, a0, false, b, (short)0, acc0, false, false);
    acc1 = __builtin_amdgcn_wmma_f32_16x16x32_bf16(false, a1, false, b, (short)0, acc1, false, false);
  }

  const int half = lane >> 4;
  const int col  = lane & 15;
#pragma unroll
  for (int v = 0; v < 8; ++v) {
    const int r    = v + 8 * half;                 // 0..15 within M-tile
    const int br0  = m0a + r;                      // global batch rows
    const int br1  = m0b + r;
    pre[gw][r][col]      = acc0[v] + xpre_t[(size_t)br0 * G_DIM + n0 + col];
    pre[gw][r + 16][col] = acc1[v] + xpre_t[(size_t)br1 * G_DIM + n0 + col];
  }
  __syncthreads();

#pragma unroll
  for (int e = tid; e < 32 * 16; e += 128) {
    const int lr = e >> 4;
    const int c  = e & 15;
    const int b    = blockIdx.y * 32 + lr;
    const int hcol = blockIdx.x * 16 + c;
    const float ig = 1.0f / (1.0f + __expf(-pre[0][lr][c]));
    const float fg = 1.0f / (1.0f + __expf(-pre[1][lr][c]));
    const float og = 1.0f / (1.0f + __expf(-pre[2][lr][c]));
    const float gg = tanhf(pre[3][lr][c]);
    const size_t idx = (size_t)b * H_DIM + hcol;
    const float ct = fg * cstate[idx] + ig * gg;
    const float ht = og * tanhf(ct);
    cstate[idx] = ct;
    hnext[idx]  = (bf16)ht;
    out_t[idx]  = ht;
    if (is_last) { hf[idx] = ht; cf[idx] = ct; }
  }
}

// ---------------------------------------------------------------------------
extern "C" void kernel_launch(void* const* d_in, const int* in_sizes, int n_in,
                              void* d_out, int out_size, void* d_ws, size_t ws_size,
                              hipStream_t stream) {
  (void)in_sizes; (void)n_in; (void)out_size; (void)ws_size;
  const float* input = (const float*)d_in[0];
  const float* h0    = (const float*)d_in[1];
  const float* c0    = (const float*)d_in[2];
  const float* W_ih  = (const float*)d_in[3];
  const float* b_ih  = (const float*)d_in[4];
  const float* W_hh  = (const float*)d_in[5];
  const float* b_hh  = (const float*)d_in[6];

  float* out = (float*)d_out;
  float* hf  = out + (size_t)T_DIM * B_DIM * H_DIM;
  float* cf  = hf + (size_t)B_DIM * H_DIM;

  char* ws = (char*)d_ws;
  size_t off = 0;
  bf16*  Xb    = (bf16*)(ws + off);  off += (size_t)T_DIM * B_DIM * I_DIM * sizeof(bf16);
  bf16*  Wihb  = (bf16*)(ws + off);  off += (size_t)G_DIM * I_DIM * sizeof(bf16);
  bf16*  Whhb  = (bf16*)(ws + off);  off += (size_t)G_DIM * H_DIM * sizeof(bf16);
  float* bias  = (float*)(ws + off); off += (size_t)G_DIM * sizeof(float);
  float* cstat = (float*)(ws + off); off += (size_t)B_DIM * H_DIM * sizeof(float);
  bf16*  hbuf0 = (bf16*)(ws + off);  off += (size_t)B_DIM * H_DIM * sizeof(bf16);
  bf16*  hbuf1 = (bf16*)(ws + off);  off += (size_t)B_DIM * H_DIM * sizeof(bf16);
  float* xpre  = (float*)(ws + off); off += (size_t)T_DIM * B_DIM * G_DIM * sizeof(float);

  // Prep: fp32 -> bf16, bias fold, state init
  cvt_f32_to_bf16<<<2048, 256, 0, stream>>>(input, Xb,   T_DIM * B_DIM * I_DIM);
  cvt_f32_to_bf16<<<1024, 256, 0, stream>>>(W_ih,  Wihb, G_DIM * I_DIM);
  cvt_f32_to_bf16<<<1024, 256, 0, stream>>>(W_hh,  Whhb, G_DIM * H_DIM);
  prep_bias_state<<<256, 256, 0, stream>>>(b_ih, b_hh, bias, h0, c0, hbuf0, cstat);

  // Phase 1: input projection GEMM (fully parallel)
  xpre_gemm<<<dim3(G_DIM / 64, (T_DIM * B_DIM) / 64), 256, 0, stream>>>(Xb, Wihb, bias, xpre);

  // Phase 2: 512 sequential fused recurrent steps (double-buffered bf16 h)
  bf16* hb[2] = { hbuf0, hbuf1 };
  for (int t = 0; t < T_DIM; ++t) {
    lstm_step<<<dim3(H_DIM / 16, 2), 128, 0, stream>>>(
        Whhb, hb[t & 1], hb[(t + 1) & 1], cstat,
        xpre + (size_t)t * B_DIM * G_DIM,
        out  + (size_t)t * B_DIM * H_DIM,
        hf, cf, (t == T_DIM - 1) ? 1 : 0);
  }
}
